// MoELayer_33852932227539
// MI455X (gfx1250) — matrix operand order repaired and checked by reading, not compiled, not run on hardware
//
#include <hip/hip_runtime.h>
#include <hip/hip_bf16.h>
#include <math.h>

// ---------------------------------------------------------------------------
// MoE layer for MI455X (gfx1250, wave32, WMMA 16x16x32 f16)
// ---------------------------------------------------------------------------
#define D_MODEL 1024
#define D_FF    4096
#define NUM_E   8
#define TOP_K   2
#define NTOK    8192
#define CAP     1280          // int(8192/8 * 1.25)
#define EROWS   (2*CAP)       // 2560 rows per expert (k=0 rows, then k=1 rows)

typedef _Float16 v16h __attribute__((ext_vector_type(16)));
typedef _Float16 v8h  __attribute__((ext_vector_type(8)));
typedef float    v8f  __attribute__((ext_vector_type(8)));
typedef float    v4f  __attribute__((ext_vector_type(4)));
typedef int      gv4i __attribute__((vector_size(16)));   // matches builtin param

// Async global->LDS copy (CDNA5 GLOBAL_LOAD_ASYNC_TO_LDS_B128), guarded so the
// file compiles on toolchains that do not expose the builtin.
#if __has_builtin(__builtin_amdgcn_global_load_async_to_lds_b128) && \
    __has_builtin(__builtin_amdgcn_s_wait_asynccnt)
#define USE_ASYNC_LDS 1
__device__ __forceinline__ void async_cp16(const void* g, void* l) {
    __builtin_amdgcn_global_load_async_to_lds_b128(
        (__attribute__((address_space(1))) gv4i*)(gv4i*)(g),
        (__attribute__((address_space(3))) gv4i*)(gv4i*)(l), 0, 0);
}
__device__ __forceinline__ void async_wait0() {
    __builtin_amdgcn_s_wait_asynccnt(0);
}
#else
#define USE_ASYNC_LDS 0
#endif

// A-matrix fragment (16x32 f16). ISA 7.12.2: lanes 0-15 hold K=0..7 & 16..23,
// lanes 16-31 hold K=8..15 & 24..31. rowp = LDS row base (32 halves per row).
__device__ __forceinline__ v16h frag_a(const _Float16* rowp, int half) {
    v8h lo = *(const v8h*)(rowp + half * 8);
    v8h hi = *(const v8h*)(rowp + 16 + half * 8);
    v16h r;
#pragma unroll
    for (int i = 0; i < 8; ++i) { r[i] = lo[i]; r[i + 8] = hi[i]; }
    return r;
}

// B-matrix fragment (32x16 f16). Lanes 0-15 hold K=0..7 & 16..23, lanes 16-31
// hold K=8..15 & 24..31 (matches A). rowp = LDS row for column n ([n][k], 32 halves/row).
__device__ __forceinline__ v16h frag_b(const _Float16* rowp, int half) {
    v8h lo = *(const v8h*)(rowp + half * 8);
    v8h hi = *(const v8h*)(rowp + 16 + half * 8);
    v16h r;
#pragma unroll
    for (int i = 0; i < 8; ++i) { r[i] = lo[i]; r[i + 8] = hi[i]; }
    return r;
}

__device__ __forceinline__ v8f wmma16(v16h a, v16h b, v8f c) {
    return __builtin_amdgcn_wmma_f32_16x16x32_f16(false, a, false, b,
                                                  (short)0, c, false, false);
}

__device__ __forceinline__ v8h cvt8(const float* p) {
    v4f f0 = *(const v4f*)(p);
    v4f f1 = *(const v4f*)(p + 4);
    v8h h;
#pragma unroll
    for (int j = 0; j < 4; ++j) { h[j] = (_Float16)f0[j]; h[j + 4] = (_Float16)f1[j]; }
    return h;
}

// ---------------------------------------------------------------------------
// 1. Router: logits -> softmax -> top2 -> normalized weights + aux stats
// ---------------------------------------------------------------------------
__global__ __launch_bounds__(256)
void moe_router(const float* __restrict__ x, const float* __restrict__ gate_w,
                int* __restrict__ top_idx, float* __restrict__ top_w,
                float* __restrict__ aux_cnt, float* __restrict__ aux_P) {
    __shared__ __align__(16) float sg[NUM_E * D_MODEL];
    int tid = threadIdx.x;
    for (int i = tid; i < NUM_E * D_MODEL; i += 256) sg[i] = gate_w[i];
    __syncthreads();

    int wave = tid >> 5, lane = tid & 31;
    int t = blockIdx.x * 8 + wave;
    const float* xr = x + (size_t)t * D_MODEL;

    float acc[NUM_E];
#pragma unroll
    for (int e = 0; e < NUM_E; ++e) acc[e] = 0.f;

#pragma unroll
    for (int c = 0; c < 8; ++c) {
        int d = c * 128 + lane * 4;
        v4f xv = *(const v4f*)(xr + d);
#pragma unroll
        for (int e = 0; e < NUM_E; ++e) {
            const float* g = sg + e * D_MODEL + d;
            acc[e] += xv[0] * g[0] + xv[1] * g[1] + xv[2] * g[2] + xv[3] * g[3];
        }
    }
#pragma unroll
    for (int off = 16; off > 0; off >>= 1)
#pragma unroll
        for (int e = 0; e < NUM_E; ++e)
            acc[e] += __shfl_xor(acc[e], off, 32);

    if (lane == 0) {
        float mx = acc[0];
#pragma unroll
        for (int e = 1; e < NUM_E; ++e) mx = fmaxf(mx, acc[e]);
        float p[NUM_E], s = 0.f;
#pragma unroll
        for (int e = 0; e < NUM_E; ++e) { p[e] = __expf(acc[e] - mx); s += p[e]; }
        float inv = 1.f / s;
#pragma unroll
        for (int e = 0; e < NUM_E; ++e) p[e] *= inv;

        int i0 = 0; float b0 = p[0];
#pragma unroll
        for (int e = 1; e < NUM_E; ++e) if (p[e] > b0) { b0 = p[e]; i0 = e; }
        int i1 = (i0 == 0) ? 1 : 0; float b1 = p[i1];
#pragma unroll
        for (int e = 0; e < NUM_E; ++e)
            if (e != i0 && p[e] > b1) { b1 = p[e]; i1 = e; }

        float ws = b0 + b1 + 1e-10f;
        top_idx[t * 2 + 0] = i0;
        top_idx[t * 2 + 1] = i1;
        top_w[t * 2 + 0] = b0 / ws;
        top_w[t * 2 + 1] = b1 / ws;
        atomicAdd(&aux_cnt[i0], 1.f);
        atomicAdd(&aux_cnt[i1], 1.f);
#pragma unroll
        for (int e = 0; e < NUM_E; ++e) atomicAdd(&aux_P[e], p[e]);
    }
}

// ---------------------------------------------------------------------------
// 2. Slot assignment: one block per (expert,k); deterministic token-order scan
// ---------------------------------------------------------------------------
__global__ __launch_bounds__(256)
void moe_assign(const int* __restrict__ top_idx, int* __restrict__ rowid) {
    int e = blockIdx.x >> 1;
    int k = blockIdx.x & 1;
    int tid = threadIdx.x;
    __shared__ int sd[256];
    int running = 0;
    for (int base = 0; base < NTOK; base += 256) {
        int t = base + tid;
        int m = (top_idx[t * 2 + k] == e) ? 1 : 0;
        sd[tid] = m;
        __syncthreads();
        for (int off = 1; off < 256; off <<= 1) {
            int v = (tid >= off) ? sd[tid - off] : 0;
            __syncthreads();
            sd[tid] += v;
            __syncthreads();
        }
        int pos = running + sd[tid] - m;  // exclusive prefix in token order
        if (m) rowid[t * 2 + k] = (pos < CAP) ? (k * CAP + pos) : -1;
        running += sd[255];
        __syncthreads();
    }
}

// ---------------------------------------------------------------------------
// 3. Gather kept tokens into per-expert f16 A buffers
// ---------------------------------------------------------------------------
__global__ __launch_bounds__(128)
void moe_gather(const float* __restrict__ x, const int* __restrict__ rowid,
                const int* __restrict__ top_idx, _Float16* __restrict__ Abuf) {
    int t = blockIdx.x, k = blockIdx.y;
    int r = rowid[t * 2 + k];
    if (r < 0) return;
    int e = top_idx[t * 2 + k];
    const float* src = x + (size_t)t * D_MODEL;
    _Float16* dst = Abuf + ((size_t)e * EROWS + r) * D_MODEL;
    int i = threadIdx.x * 8;
    *(v8h*)(dst + i) = cvt8(src + i);
}

// ---------------------------------------------------------------------------
// 4a. GEMM13: H = silu(A @ W1^T) * (A @ W3^T)   [2560 x 4096] f16
//     Double-buffered LDS; global loads for chunk i+1 overlap WMMAs of chunk i.
// ---------------------------------------------------------------------------
__global__ __launch_bounds__(256)
void moe_gemm13(const _Float16* __restrict__ A, const float* __restrict__ W1,
                const float* __restrict__ W3, _Float16* __restrict__ H) {
    __shared__ __align__(16) _Float16 sA[2][128 * 32];
    __shared__ __align__(16) _Float16 sB1[2][64 * 32];
    __shared__ __align__(16) _Float16 sB3[2][64 * 32];

    int tid = threadIdx.x;
    int nBase = blockIdx.x * 64;
    int rBase = blockIdx.y * 128;
    int wave = tid >> 5, lane = tid & 31;

    v8f acc1[4], acc3[4];
#pragma unroll
    for (int i = 0; i < 4; ++i)
#pragma unroll
        for (int v = 0; v < 8; ++v) { acc1[i][v] = 0.f; acc3[i][v] = 0.f; }

    int ar = tid >> 1, aseg = tid & 1;  // A copy: 128 rows x 2 segs of 16 halves
    int br = tid >> 2, bq = tid & 3;    // B copy: 64 rows x 4 segs of 8 f32

    const _Float16* aSrcBase = A + (size_t)(rBase + ar) * D_MODEL + aseg * 16;
    const float*    b1Base   = W1 + (size_t)(nBase + br) * D_MODEL + bq * 8;
    const float*    b3Base   = W3 + (size_t)(nBase + br) * D_MODEL + bq * 8;
    _Float16* aDst0 = &sA[0][ar * 32 + aseg * 16];
    _Float16* aDst1 = &sA[1][ar * 32 + aseg * 16];
    _Float16* b1Dst0 = &sB1[0][br * 32 + bq * 8];
    _Float16* b1Dst1 = &sB1[1][br * 32 + bq * 8];
    _Float16* b3Dst0 = &sB3[0][br * 32 + bq * 8];
    _Float16* b3Dst1 = &sB3[1][br * 32 + bq * 8];

    // ---- stage chunk 0 into buffer 0
#if USE_ASYNC_LDS
    async_cp16(aSrcBase, aDst0);
    async_cp16(aSrcBase + 8, aDst0 + 8);
#else
    { const uint4* s = (const uint4*)aSrcBase;
      ((uint4*)aDst0)[0] = s[0]; ((uint4*)aDst0)[1] = s[1]; }
#endif
    *(v8h*)b1Dst0 = cvt8(b1Base);
    *(v8h*)b3Dst0 = cvt8(b3Base);
#if USE_ASYNC_LDS
    async_wait0();
#endif
    __syncthreads();

    const int NIT = D_MODEL / 32;
    for (int it = 0; it < NIT; ++it) {
        int cur = it & 1;
        bool hasNext = (it + 1) < NIT;
        int knext = (it + 1) * 32;

        // ---- issue next-chunk global loads (overlap with WMMAs below)
        uint4 aR0, aR1; v8h b1R, b3R;
        if (hasNext) {
#if USE_ASYNC_LDS
            _Float16* ad = cur ? aDst0 : aDst1;
            async_cp16(aSrcBase + knext, ad);
            async_cp16(aSrcBase + knext + 8, ad + 8);
#else
            const uint4* s = (const uint4*)(aSrcBase + knext);
            aR0 = s[0]; aR1 = s[1];
#endif
            b1R = cvt8(b1Base + knext);
            b3R = cvt8(b3Base + knext);
        }

        // ---- compute from current buffer: batch fragment loads, then WMMAs
        int m = lane & 15, half = lane >> 4;
        const _Float16* aT  = &sA[cur][0];
        const _Float16* b1T = &sB1[cur][0];
        const _Float16* b3T = &sB3[cur][0];
        v16h a = frag_a(aT + (wave * 16 + m) * 32, half);
        v16h bf[4];
#pragma unroll
        for (int ns = 0; ns < 4; ++ns) bf[ns] = frag_b(b1T + (ns * 16 + m) * 32, half);
#pragma unroll
        for (int ns = 0; ns < 4; ++ns) acc1[ns] = wmma16(a, bf[ns], acc1[ns]);
#pragma unroll
        for (int ns = 0; ns < 4; ++ns) bf[ns] = frag_b(b3T + (ns * 16 + m) * 32, half);
#pragma unroll
        for (int ns = 0; ns < 4; ++ns) acc3[ns] = wmma16(a, bf[ns], acc3[ns]);

        // ---- commit next chunk to the other buffer
        if (hasNext) {
#if !USE_ASYNC_LDS
            uint4* ad = (uint4*)(cur ? aDst0 : aDst1);
            ad[0] = aR0; ad[1] = aR1;
#endif
            *(v8h*)(cur ? b1Dst0 : b1Dst1) = b1R;
            *(v8h*)(cur ? b3Dst0 : b3Dst1) = b3R;
#if USE_ASYNC_LDS
            async_wait0();
#endif
        }
        __syncthreads();
    }

    // Epilogue: silu(s1)*s3 -> f16. C/D layout: lane = (m/8)*16 + n, vgpr = m%8
    int nl = lane & 15, rs = lane >> 4;
#pragma unroll
    for (int ns = 0; ns < 4; ++ns)
#pragma unroll
        for (int v = 0; v < 8; ++v) {
            float s1 = acc1[ns][v], s3 = acc3[ns][v];
            float hv = s1 * __builtin_amdgcn_rcpf(1.f + __expf(-s1)) * s3;
            size_t gr = rBase + wave * 16 + rs * 8 + v;
            size_t gc = nBase + ns * 16 + nl;
            H[gr * D_FF + gc] = (_Float16)hv;
        }
}

// ---------------------------------------------------------------------------
// 4b. GEMM2: Out = H @ W2^T   [2560 x 1024] f32, same pipeline structure
// ---------------------------------------------------------------------------
__global__ __launch_bounds__(256)
void moe_gemm2(const _Float16* __restrict__ H, const float* __restrict__ W2,
               float* __restrict__ Out) {
    __shared__ __align__(16) _Float16 sA[2][128 * 32];
    __shared__ __align__(16) _Float16 sB[2][64 * 32];

    int tid = threadIdx.x;
    int nBase = blockIdx.x * 64;
    int rBase = blockIdx.y * 128;
    int wave = tid >> 5, lane = tid & 31;

    v8f acc[4];
#pragma unroll
    for (int i = 0; i < 4; ++i)
#pragma unroll
        for (int v = 0; v < 8; ++v) acc[i][v] = 0.f;

    int ar = tid >> 1, aseg = tid & 1;
    int br = tid >> 2, bq = tid & 3;

    const _Float16* aSrcBase = H + (size_t)(rBase + ar) * D_FF + aseg * 16;
    const float*    bBase    = W2 + (size_t)(nBase + br) * D_FF + bq * 8;
    _Float16* aDst0 = &sA[0][ar * 32 + aseg * 16];
    _Float16* aDst1 = &sA[1][ar * 32 + aseg * 16];
    _Float16* bDst0 = &sB[0][br * 32 + bq * 8];
    _Float16* bDst1 = &sB[1][br * 32 + bq * 8];

#if USE_ASYNC_LDS
    async_cp16(aSrcBase, aDst0);
    async_cp16(aSrcBase + 8, aDst0 + 8);
#else
    { const uint4* s = (const uint4*)aSrcBase;
      ((uint4*)aDst0)[0] = s[0]; ((uint4*)aDst0)[1] = s[1]; }
#endif
    *(v8h*)bDst0 = cvt8(bBase);
#if USE_ASYNC_LDS
    async_wait0();
#endif
    __syncthreads();

    const int NIT = D_FF / 32;
    for (int it = 0; it < NIT; ++it) {
        int cur = it & 1;
        bool hasNext = (it + 1) < NIT;
        int knext = (it + 1) * 32;

        uint4 aR0, aR1; v8h bR;
        if (hasNext) {
#if USE_ASYNC_LDS
            _Float16* ad = cur ? aDst0 : aDst1;
            async_cp16(aSrcBase + knext, ad);
            async_cp16(aSrcBase + knext + 8, ad + 8);
#else
            const uint4* s = (const uint4*)(aSrcBase + knext);
            aR0 = s[0]; aR1 = s[1];
#endif
            bR = cvt8(bBase + knext);
        }

        int m = lane & 15, half = lane >> 4;
        const _Float16* aT = &sA[cur][0];
        const _Float16* bT = &sB[cur][0];
        v16h a = frag_a(aT + (wave * 16 + m) * 32, half);
        v16h bf[4];
#pragma unroll
        for (int ns = 0; ns < 4; ++ns) bf[ns] = frag_b(bT + (ns * 16 + m) * 32, half);
#pragma unroll
        for (int ns = 0; ns < 4; ++ns) acc[ns] = wmma16(a, bf[ns], acc[ns]);

        if (hasNext) {
#if !USE_ASYNC_LDS
            uint4* ad = (uint4*)(cur ? aDst0 : aDst1);
            ad[0] = aR0; ad[1] = aR1;
#endif
            *(v8h*)(cur ? bDst0 : bDst1) = bR;
#if USE_ASYNC_LDS
            async_wait0();
#endif
        }
        __syncthreads();
    }

    int nl = lane & 15, rs = lane >> 4;
#pragma unroll
    for (int ns = 0; ns < 4; ++ns)
#pragma unroll
        for (int v = 0; v < 8; ++v) {
            size_t gr = rBase + wave * 16 + rs * 8 + v;
            size_t gc = nBase + ns * 16 + nl;
            Out[gr * D_MODEL + gc] = acc[ns][v];
        }
}

// ---------------------------------------------------------------------------
// 5. Combine + aux loss
// ---------------------------------------------------------------------------
__global__ __launch_bounds__(256)
void moe_combine(const float* __restrict__ OutBuf, const int* __restrict__ rowid,
                 const int* __restrict__ top_idx, const float* __restrict__ top_w,
                 float* __restrict__ out) {
    int t = blockIdx.x;
    int c = threadIdx.x * 4;
    v4f acc; acc[0] = 0.f; acc[1] = 0.f; acc[2] = 0.f; acc[3] = 0.f;
#pragma unroll
    for (int k = 0; k < TOP_K; ++k) {
        int r = rowid[t * 2 + k];
        if (r >= 0) {
            int e = top_idx[t * 2 + k];
            float w = top_w[t * 2 + k];
            v4f v = *(const v4f*)(OutBuf + ((size_t)e * EROWS + r) * D_MODEL + c);
#pragma unroll
            for (int j = 0; j < 4; ++j) acc[j] += w * v[j];
        }
    }
    *(v4f*)(out + (size_t)t * D_MODEL + c) = acc;
}

__global__ void moe_finalize_aux(const float* __restrict__ cnt,
                                 const float* __restrict__ P,
                                 float* __restrict__ out_aux) {
    if (threadIdx.x == 0 && blockIdx.x == 0) {
        float s = 0.f;
#pragma unroll
        for (int e = 0; e < NUM_E; ++e)
            s += (cnt[e] / (float)(NTOK * TOP_K)) * (P[e] / (float)NTOK);
        out_aux[0] = (float)NUM_E * s;
    }
}

// ---------------------------------------------------------------------------
// Launch
// ---------------------------------------------------------------------------
extern "C" void kernel_launch(void* const* d_in, const int* in_sizes, int n_in,
                              void* d_out, int out_size, void* d_ws, size_t ws_size,
                              hipStream_t stream) {
    const float* x      = (const float*)d_in[0];
    const float* gate_w = (const float*)d_in[1];
    const float* w1     = (const float*)d_in[2];
    const float* w2     = (const float*)d_in[3];
    const float* w3     = (const float*)d_in[4];
    float* out = (float*)d_out;

    char* ws = (char*)d_ws;
    float*    aux_cnt = (float*)(ws + 0);       // 8 f32
    float*    aux_P   = (float*)(ws + 32);      // 8 f32
    int*      top_idx = (int*)(ws + 256);                     // 8192*2
    float*    top_w   = (float*)(ws + 256 + 65536);           // 8192*2
    int*      rowid   = (int*)(ws + 256 + 131072);            // 8192*2
    _Float16* Abuf    = (_Float16*)(ws + 196864);             // 8*2560*1024 f16
    _Float16* Hbuf    = (_Float16*)(ws + 42139904);           // 2560*4096 f16
    float*    OutBuf  = (float*)(ws + 63111424);              // 8*2560*1024 f32

    (void)hipMemsetAsync(ws, 0, 64, stream);                  // aux accumulators
    (void)hipMemsetAsync(Abuf, 0, (size_t)NUM_E * EROWS * D_MODEL * 2, stream);

    moe_router<<<NTOK / 8, 256, 0, stream>>>(x, gate_w, top_idx, top_w, aux_cnt, aux_P);
    moe_assign<<<NUM_E * TOP_K, 256, 0, stream>>>(top_idx, rowid);
    moe_gather<<<dim3(NTOK, TOP_K), 128, 0, stream>>>(x, rowid, top_idx, Abuf);

    for (int e = 0; e < NUM_E; ++e) {
        const _Float16* Ae = Abuf + (size_t)e * EROWS * D_MODEL;
        const float* W1e = w1 + (size_t)e * D_FF * D_MODEL;
        const float* W3e = w3 + (size_t)e * D_FF * D_MODEL;
        const float* W2e = w2 + (size_t)e * D_MODEL * D_FF;
        float* Oe = OutBuf + (size_t)e * EROWS * D_MODEL;
        moe_gemm13<<<dim3(D_FF / 64, EROWS / 128), 256, 0, stream>>>(Ae, W1e, W3e, Hbuf);
        moe_gemm2<<<dim3(D_MODEL / 64, EROWS / 128), 256, 0, stream>>>(Hbuf, W2e, Oe);
    }

    moe_combine<<<NTOK, 256, 0, stream>>>(OutBuf, rowid, top_idx, top_w, out);
    moe_finalize_aux<<<1, 32, 0, stream>>>(aux_cnt, aux_P, out + (size_t)NTOK * D_MODEL);
}
